// MultiHeadedAttention_61589831024845
// MI455X (gfx1250) — matrix operand order repaired and checked by reading, not compiled, not run on hardware
//
#include <hip/hip_runtime.h>
#include <hip/hip_bf16.h>

typedef __attribute__((ext_vector_type(16))) _Float16 v16h;
typedef __attribute__((ext_vector_type(8)))  _Float16 v8h;
typedef __attribute__((ext_vector_type(8)))  float    v8f;
typedef __attribute__((ext_vector_type(4)))  float    v4f;
typedef int v4i __attribute__((vector_size(16)));   // matches builtin param pointee

#define AS1 __attribute__((address_space(1)))
#define AS3 __attribute__((address_space(3)))

#if __has_builtin(__builtin_amdgcn_global_load_async_to_lds_b128)
#define ATTN_ASYNC 1
#else
#define ATTN_ASYNC 0
#endif

constexpr int H_     = 8;
constexpr int DM     = 512;
constexpr int DK     = 64;
constexpr int LQ     = 4096;
constexpr int LK     = 4096;
constexpr int NB     = 2;
constexpr int MROWS  = NB * LQ;      // 8192 total sequence rows

// ---------------- fragment loaders (wave32; contiguous k = 16*(lane/16) + s, same in A and B) ----
// A-frag 16x32 f16: caller passes pointer to row (lane&15)'s data.
// element h -> k = (lane>=16?16:0) + h  (16 contiguous)
static __device__ __forceinline__ v16h load_a_frag(const _Float16* rowp, int lane) {
  const _Float16* p = rowp + ((lane & 16) ? 16 : 0);
  v8h c0 = *(const v8h*)(p);
  v8h c1 = *(const v8h*)(p + 8);
  v16h a;
#pragma unroll
  for (int i = 0; i < 8; ++i) { a[i] = c0[i]; a[i + 8] = c1[i]; }
  return a;
}
// B-frag 32x16 f16: caller passes pointer to column n=lane&15's k-major data.
// element h -> k = (lane>=16?16:0) + h  (16 contiguous)
static __device__ __forceinline__ v16h load_b_frag(const _Float16* colp, int lane) {
  const _Float16* p = colp + ((lane & 16) ? 16 : 0);
  v8h c0 = *(const v8h*)(p);
  v8h c1 = *(const v8h*)(p + 8);
  v16h b;
#pragma unroll
  for (int i = 0; i < 8; ++i) { b[i] = c0[i]; b[i + 8] = c1[i]; }
  return b;
}

static __device__ __forceinline__ void wait_async0() {
#if __has_builtin(__builtin_amdgcn_s_wait_asynccnt)
  __builtin_amdgcn_s_wait_asynccnt(0);
#else
  asm volatile("s_wait_asynccnt 0" ::: "memory");
#endif
}

static __device__ __forceinline__ void async_cp16(const _Float16* g, _Float16* l) {
#if ATTN_ASYNC
  __builtin_amdgcn_global_load_async_to_lds_b128((AS1 v4i*)(g), (AS3 v4i*)(l), 0, 0);
#else
  (void)g; (void)l;
#endif
}

// ---------------- small prep kernels --------------------------------------
__global__ void f32_to_f16_kernel(const float* __restrict__ in,
                                  _Float16* __restrict__ out, int n) {
  int i = blockIdx.x * blockDim.x + threadIdx.x;
  if (i < n) out[i] = (_Float16)in[i];
}

__global__ void mask_bias_kernel(const int* __restrict__ mask,
                                 float* __restrict__ bias, int n) {
  int i = blockIdx.x * blockDim.x + threadIdx.x;
  if (i < n) bias[i] = (mask[i] == 0) ? -1000000000.0f : 0.0f;
}

// ---------------- Q/K projection: Yt = W * X^T (transposed output) --------
// dst layout [NB][H][L][DK] f16 ; scale folds 1/sqrt(dk) into Q.
__global__ void proj_qk_kernel(const _Float16* __restrict__ Xh,   // [MROWS][DM]
                               const _Float16* __restrict__ Wh,   // [DM][DM]
                               const float*    __restrict__ bias, // [DM]
                               _Float16*       __restrict__ dst,
                               float scale) {
  const int lane = threadIdx.x & 31;
  const int wave = threadIdx.x >> 5;
  const int tile = blockIdx.x * (blockDim.x >> 5) + wave;
  const int FT = DM / 16;                    // 32 feature tiles
  const int f0 = (tile % FT) * 16;
  const int m0 = (tile / FT) * 16;
  const int hs8 = (lane & 16) ? 8 : 0;

  const _Float16* arow = Wh + (size_t)(f0 + (lane & 15)) * DM;
  const _Float16* brow = Xh + (size_t)(m0 + (lane & 15)) * DM;
  v8f acc = {};
#pragma unroll 4
  for (int k = 0; k < DM; k += 32) {
    v16h a = load_a_frag(arow + k, lane);
    v16h b = load_b_frag(brow + k, lane);
    acc = __builtin_amdgcn_wmma_f32_16x16x32_f16(false, a, false, b,
                                                 (short)0, acc, false, false);
  }
  const int mg  = m0 + (lane & 15);
  const int bb  = mg / LQ;
  const int seq = mg % LQ;
  const int f   = f0 + hs8;                  // first of 8 consecutive features
  const int hh  = f / DK;
  const int d   = f % DK;
  v8h outv;
#pragma unroll
  for (int r = 0; r < 8; ++r)
    outv[r] = (_Float16)((acc[r] + bias[f + r]) * scale);
  *(v8h*)(dst + (((size_t)(bb * H_ + hh) * LQ + seq) * DK + d)) = outv;
}

// ---------------- V projection: Y = X * W^T, stored transposed per head ----
// vt layout [NB][H][DK][LK] f16
__global__ void proj_v_kernel(const _Float16* __restrict__ Xh,
                              const _Float16* __restrict__ Wh,
                              const float*    __restrict__ bias,
                              _Float16*       __restrict__ vt) {
  const int lane = threadIdx.x & 31;
  const int wave = threadIdx.x >> 5;
  const int tile = blockIdx.x * (blockDim.x >> 5) + wave;
  const int FT = DM / 16;
  const int f0 = (tile % FT) * 16;
  const int m0 = (tile / FT) * 16;
  const int hs8 = (lane & 16) ? 8 : 0;

  const _Float16* arow = Xh + (size_t)(m0 + (lane & 15)) * DM;
  const _Float16* brow = Wh + (size_t)(f0 + (lane & 15)) * DM; // B[k][n]=W[n][k]
  v8f acc = {};
#pragma unroll 4
  for (int k = 0; k < DM; k += 32) {
    v16h a = load_a_frag(arow + k, lane);
    v16h b = load_b_frag(brow + k, lane);
    acc = __builtin_amdgcn_wmma_f32_16x16x32_f16(false, a, false, b,
                                                 (short)0, acc, false, false);
  }
  const int f   = f0 + (lane & 15);
  const int hh  = f / DK;
  const int d   = f % DK;
  const int mg  = m0 + hs8;                  // first of 8 consecutive seq rows
  const int bb  = mg / LQ;
  const int seq = mg % LQ;
  const float bf = bias[f];
  v8h outv;
#pragma unroll
  for (int r = 0; r < 8; ++r) outv[r] = (_Float16)(acc[r] + bf);
  *(v8h*)(vt + (((size_t)(bb * H_ + hh) * DK + d) * LK + seq)) = outv;
}

// ---------------- flash attention -----------------------------------------
// 8 waves/block share (b,h); K/V tiles staged once per block into LDS.
__global__ void __launch_bounds__(256)
attn_kernel(const _Float16* __restrict__ qh,   // [NB][H][LQ][DK], pre-scaled
            const _Float16* __restrict__ kh,   // [NB][H][LK][DK]
            const _Float16* __restrict__ vt,   // [NB][H][DK][LK]
            const float*    __restrict__ bias, // [NB][LK]
            _Float16*       __restrict__ ctx)  // [MROWS][DM]
{
  __shared__ __align__(16) _Float16 kbuf[2][32 * DK];  // [key][d]   4KB each
  __shared__ __align__(16) _Float16 vbuf[2][DK * 32];  // [d][key]   4KB each

  const int tid  = threadIdx.x;
  const int lane = tid & 31;
  const int wave = tid >> 5;
  const int wid  = blockIdx.x * 8 + wave;
  const int QT = LQ / 16;                    // 256 q-tiles per head
  const int q0 = (wid % QT) * 16;
  const int bh = wid / QT;                   // uniform across block (256%8==0)
  const int bb = bh / H_;
  const int hh = bh % H_;
  const int hs = (lane >> 4) & 1;

  const _Float16* krow = kh + (size_t)bh * LK * DK;
  const _Float16* vrow = vt + (size_t)bh * DK * LK;
  const float*    bp   = bias + (size_t)bb * LK;

  // per-thread copy of chunk c (0..511, 16B each): c<256 -> K tile, else V tile
  auto chunk_ptrs = [&](int c, int kb, int buf,
                        const _Float16*& g, _Float16*& l) {
    if (c < 256) {
      const int key = c >> 3, dp = (c & 7) * 8;
      g = krow + (size_t)(kb + key) * DK + dp;
      l = &kbuf[buf][key * DK + dp];
    } else {
      const int c2 = c - 256;
      const int d = c2 >> 2, kp = (c2 & 3) * 8;
      g = vrow + (size_t)d * LK + kb + kp;
      l = &vbuf[buf][d * 32 + kp];
    }
  };

  // Preload Q as two B-fragments (d = 0..31 and 32..63)
  const _Float16* qp = qh + ((size_t)bh * LQ + q0 + (lane & 15)) * DK + (hs ? 16 : 0);
  v16h qb0, qb1;
  {
    v8h c0 = *(const v8h*)(qp);      v8h c1 = *(const v8h*)(qp + 8);
    v8h c2 = *(const v8h*)(qp + 32); v8h c3 = *(const v8h*)(qp + 40);
#pragma unroll
    for (int i = 0; i < 8; ++i) {
      qb0[i] = c0[i]; qb0[i + 8] = c1[i];
      qb1[i] = c2[i]; qb1[i + 8] = c3[i];
    }
  }

  // stage tile 0 into buffer 0
  {
    const _Float16 *g0, *g1; _Float16 *l0, *l1;
    chunk_ptrs(2 * tid, 0, 0, g0, l0);
    chunk_ptrs(2 * tid + 1, 0, 0, g1, l1);
#if ATTN_ASYNC
    async_cp16(g0, l0);
    async_cp16(g1, l1);
    wait_async0();
#else
    *(v8h*)l0 = *(const v8h*)g0;
    *(v8h*)l1 = *(const v8h*)g1;
#endif
  }
  __syncthreads();

  float mprev = -1e30f;
  float lsum  = 0.0f;
  v8f o[4] = {};

  constexpr int NIT = LK / 32;               // 128 key blocks
  for (int it = 0; it < NIT; ++it) {
    const int kb  = it * 32;
    const int cur = it & 1, nxt = cur ^ 1;
    const bool havenext = (it + 1 < NIT);

    const _Float16 *g0 = nullptr, *g1 = nullptr;
    _Float16 *l0 = nullptr, *l1 = nullptr;
#if !ATTN_ASYNC
    v8h r0 = {}, r1 = {};
#endif
    if (havenext) {
      chunk_ptrs(2 * tid, kb + 32, nxt, g0, l0);
      chunk_ptrs(2 * tid + 1, kb + 32, nxt, g1, l1);
#if ATTN_ASYNC
      // safe: barrier at end of previous iteration retired all reads of buf[nxt]
      async_cp16(g0, l0);
      async_cp16(g1, l1);
#else
      r0 = *(const v8h*)g0;                  // latency overlaps compute below
      r1 = *(const v8h*)g1;
#endif
    }

    // ---- S^T tiles from LDS: D[m=key][n=query] = sum_d K[key][d]*Q[query][d]
    const _Float16* k0row = &kbuf[cur][(lane & 15) * DK];
    const _Float16* k1row = k0row + 16 * DK;
    v8f s0 = {}, s1 = {};
    {
      v16h a = load_a_frag(k0row, lane);
      s0 = __builtin_amdgcn_wmma_f32_16x16x32_f16(false, a, false, qb0, (short)0, s0, false, false);
      a = load_a_frag(k0row + 32, lane);
      s0 = __builtin_amdgcn_wmma_f32_16x16x32_f16(false, a, false, qb1, (short)0, s0, false, false);
      a = load_a_frag(k1row, lane);
      s1 = __builtin_amdgcn_wmma_f32_16x16x32_f16(false, a, false, qb0, (short)0, s1, false, false);
      a = load_a_frag(k1row + 32, lane);
      s1 = __builtin_amdgcn_wmma_f32_16x16x32_f16(false, a, false, qb1, (short)0, s1, false, false);
    }
    // additive mask bias: s0 keys = kb+hs*8+r, s1 keys = kb+16+hs*8+r
    const float* mb = bp + kb + hs * 8;
    v4f m00 = *(const v4f*)(mb);      v4f m01 = *(const v4f*)(mb + 4);
    v4f m10 = *(const v4f*)(mb + 16); v4f m11 = *(const v4f*)(mb + 20);
    float mloc = -1e30f;
#pragma unroll
    for (int r = 0; r < 4; ++r) {
      s0[r] += m00[r]; s0[r + 4] += m01[r];
      s1[r] += m10[r]; s1[r + 4] += m11[r];
    }
#pragma unroll
    for (int r = 0; r < 8; ++r) mloc = fmaxf(mloc, fmaxf(s0[r], s1[r]));
    mloc = fmaxf(mloc, __shfl_xor(mloc, 16, 32));
    const float mnew  = fmaxf(mprev, mloc);
    const float alpha = __expf(mprev - mnew);
    float rsum = 0.0f;
    v16h pf;                                   // P as A-fragment directly
#pragma unroll
    for (int r = 0; r < 8; ++r) {
      float p0 = __expf(s0[r] - mnew);
      float p1 = __expf(s1[r] - mnew);
      rsum += p0 + p1;
      pf[r]     = (_Float16)p0;
      pf[r + 8] = (_Float16)p1;
    }
    rsum += __shfl_xor(rsum, 16, 32);
    lsum = lsum * alpha + rsum;
    mprev = mnew;
    // rescale accumulators: alpha for query row (r + hs*8) lives in lane (r + hs*8)
#pragma unroll
    for (int r = 0; r < 8; ++r) {
      float ar = __shfl(alpha, r + hs * 8, 32);
      o[0][r] *= ar; o[1][r] *= ar; o[2][r] *= ar; o[3][r] *= ar;
    }
    // O += P @ V  (B-frags from LDS V^T tile: keys hs*8+i and 16+hs*8+i per lane)
#pragma unroll
    for (int dt = 0; dt < 4; ++dt) {
      const _Float16* vp = &vbuf[cur][(dt * 16 + (lane & 15)) * 32 + hs * 8];
      v8h c0 = *(const v8h*)(vp); v8h c1 = *(const v8h*)(vp + 16);
      v16h vb;
#pragma unroll
      for (int i = 0; i < 8; ++i) { vb[i] = c0[i]; vb[i + 8] = c1[i]; }
      o[dt] = __builtin_amdgcn_wmma_f32_16x16x32_f16(false, pf, false, vb,
                                                     (short)0, o[dt], false, false);
    }

    if (havenext) {
#if ATTN_ASYNC
      wait_async0();                          // buf[nxt] writes complete
#else
      *(v8h*)l0 = r0;                         // commit staged tile
      *(v8h*)l1 = r1;
#endif
      __syncthreads();                        // visible to all waves
    }
  }

  // normalize and scatter to ctx[b*LQ+seq][h*64 + d]
  const float linv = 1.0f / lsum;              // lane holds lsum of query lane&15
  const int fcol = hh * DK + (lane & 15);
#pragma unroll
  for (int r = 0; r < 8; ++r) {
    float lr = __shfl(linv, r + hs * 8, 32);
    size_t row = (size_t)(bb * LQ + q0 + hs * 8 + r) * DM;
    ctx[row + fcol +  0] = (_Float16)(o[0][r] * lr);
    ctx[row + fcol + 16] = (_Float16)(o[1][r] * lr);
    ctx[row + fcol + 32] = (_Float16)(o[2][r] * lr);
    ctx[row + fcol + 48] = (_Float16)(o[3][r] * lr);
  }
}

// ---------------- output projection: out = ctx * Wo^T + bo (fp32 out) ------
__global__ void proj_o_kernel(const _Float16* __restrict__ Ctx,  // [MROWS][DM]
                              const _Float16* __restrict__ Wh,   // Wo [DM][DM]
                              const float*    __restrict__ bias, // [DM]
                              float*          __restrict__ out)  // [MROWS][DM]
{
  const int lane = threadIdx.x & 31;
  const int wave = threadIdx.x >> 5;
  const int tile = blockIdx.x * (blockDim.x >> 5) + wave;
  const int FT = DM / 16;
  const int f0 = (tile % FT) * 16;
  const int m0 = (tile / FT) * 16;
  const int hs8 = (lane & 16) ? 8 : 0;

  const _Float16* arow = Wh  + (size_t)(f0 + (lane & 15)) * DM;
  const _Float16* brow = Ctx + (size_t)(m0 + (lane & 15)) * DM;
  v8f acc = {};
#pragma unroll 4
  for (int k = 0; k < DM; k += 32) {
    v16h a = load_a_frag(arow + k, lane);
    v16h b = load_b_frag(brow + k, lane);
    acc = __builtin_amdgcn_wmma_f32_16x16x32_f16(false, a, false, b,
                                                 (short)0, acc, false, false);
  }
  const int mg = m0 + (lane & 15);
  const int f  = f0 + hs8;
  float* op = out + (size_t)mg * DM + f;
  v4f o0, o1;
#pragma unroll
  for (int r = 0; r < 4; ++r) {
    o0[r] = acc[r]     + bias[f + r];
    o1[r] = acc[r + 4] + bias[f + 4 + r];
  }
  *(v4f*)(op)     = o0;
  *(v4f*)(op + 4) = o1;
}

// ---------------- host launcher -------------------------------------------
extern "C" void kernel_launch(void* const* d_in, const int* in_sizes, int n_in,
                              void* d_out, int out_size, void* d_ws, size_t ws_size,
                              hipStream_t stream) {
  (void)in_sizes; (void)n_in; (void)out_size; (void)ws_size;
  const float* q    = (const float*)d_in[0];
  const float* k    = (const float*)d_in[1];
  const float* v    = (const float*)d_in[2];
  const int*   mask = (const int*)d_in[3];
  const float* Wq   = (const float*)d_in[4];
  const float* bq   = (const float*)d_in[5];
  const float* Wk   = (const float*)d_in[6];
  const float* bk   = (const float*)d_in[7];
  const float* Wv   = (const float*)d_in[8];
  const float* bv   = (const float*)d_in[9];
  const float* Wo   = (const float*)d_in[10];
  const float* bo   = (const float*)d_in[11];
  float* out = (float*)d_out;

  char* ws = (char*)d_ws;
  size_t off = 0;
  auto carve = [&](size_t bytes) -> char* {
    char* p = ws + off;
    off += (bytes + 255) & ~(size_t)255;
    return p;
  };
  const size_t XE = (size_t)MROWS * DM;         // 4,194,304 elements
  const size_t WE = (size_t)DM * DM;
  _Float16* xqh = (_Float16*)carve(XE * 2);
  _Float16* xkh = (_Float16*)carve(XE * 2);
  _Float16* xvh = (_Float16*)carve(XE * 2);
  _Float16* wqh = (_Float16*)carve(WE * 2);
  _Float16* wkh = (_Float16*)carve(WE * 2);
  _Float16* wvh = (_Float16*)carve(WE * 2);
  _Float16* woh = (_Float16*)carve(WE * 2);
  _Float16* qhd = (_Float16*)carve(XE * 2);     // [NB][H][LQ][DK]
  _Float16* khd = (_Float16*)carve(XE * 2);     // [NB][H][LK][DK]
  _Float16* vtd = (_Float16*)carve(XE * 2);     // [NB][H][DK][LK]
  _Float16* ctx = (_Float16*)carve(XE * 2);     // [MROWS][DM]
  float*    mbias = (float*)carve((size_t)NB * LK * 4);

  // precision conversion
  f32_to_f16_kernel<<<(int)(XE / 256), 256, 0, stream>>>(q, xqh, (int)XE);
  f32_to_f16_kernel<<<(int)(XE / 256), 256, 0, stream>>>(k, xkh, (int)XE);
  f32_to_f16_kernel<<<(int)(XE / 256), 256, 0, stream>>>(v, xvh, (int)XE);
  f32_to_f16_kernel<<<(int)(WE / 256), 256, 0, stream>>>(Wq, wqh, (int)WE);
  f32_to_f16_kernel<<<(int)(WE / 256), 256, 0, stream>>>(Wk, wkh, (int)WE);
  f32_to_f16_kernel<<<(int)(WE / 256), 256, 0, stream>>>(Wv, wvh, (int)WE);
  f32_to_f16_kernel<<<(int)(WE / 256), 256, 0, stream>>>(Wo, woh, (int)WE);
  mask_bias_kernel<<<(NB * LK) / 256, 256, 0, stream>>>(mask, mbias, NB * LK);

  const int ptiles = (DM / 16) * (MROWS / 16);  // 16384 tiles, 4 waves/block
  proj_qk_kernel<<<ptiles / 4, 128, 0, stream>>>(xqh, wqh, bq, qhd, 0.125f);
  proj_qk_kernel<<<ptiles / 4, 128, 0, stream>>>(xkh, wkh, bk, khd, 1.0f);
  proj_v_kernel <<<ptiles / 4, 128, 0, stream>>>(xvh, wvh, bv, vtd);

  const int awaves = NB * H_ * (LQ / 16);       // 4096 waves, 8 waves/block
  attn_kernel<<<awaves / 8, 256, 0, stream>>>(qhd, khd, vtd, mbias, ctx);

  proj_o_kernel<<<ptiles / 4, 128, 0, stream>>>(ctx, woh, bo, out);
}
